// BidirectionalPropagation_10866267259445
// MI455X (gfx1250) — matrix operand order repaired and checked by reading, not compiled, not run on hardware
//
#include <hip/hip_runtime.h>
#include <hip/hip_bf16.h>

// ---------------------------------------------------------------------------
// Types for CDNA5 WMMA (wave32): A/B = 16 bf16 per lane (8 VGPRs), C/D = 8 f32.
// ---------------------------------------------------------------------------
typedef __attribute__((ext_vector_type(16))) __bf16        v16bf;
typedef __attribute__((ext_vector_type(8)))  float         v8f;
typedef __attribute__((ext_vector_type(8)))  unsigned int  v8u;

#define HW   4096
#define CH   128
#define CHW  (CH * HW)   // 524288

__device__ __forceinline__ unsigned short f2bf(float f) {
  unsigned int u = __float_as_uint(f);
  u += 0x7fffu + ((u >> 16) & 1u);          // round-to-nearest-even
  return (unsigned short)(u >> 16);
}

// ---------------------------------------------------------------------------
// fp32 -> bf16 pack (flat)
// ---------------------------------------------------------------------------
__global__ __launch_bounds__(256) void pack_bf16_k(const float* __restrict__ src,
                                                   unsigned short* __restrict__ dst,
                                                   int n) {
  int i = blockIdx.x * 256 + threadIdx.x;
  if (i < n) dst[i] = f2bf(src[i]);
}

// ---------------------------------------------------------------------------
// fp32 OIHW [Cout][Cin][3][3] -> bf16 tap-major [Cout][tap*Cin + ci]
// (lets the GEMM K dimension keep a single 3x3 tap per 32-wide K block)
// ---------------------------------------------------------------------------
__global__ __launch_bounds__(256) void pack_w_tap_k(const float* __restrict__ src,
                                                    unsigned short* __restrict__ dst,
                                                    int Cout, int Cin) {
  int i = blockIdx.x * 256 + threadIdx.x;
  int total = Cout * Cin * 9;
  if (i >= total) return;
  int co  = i / (Cin * 9);
  int rem = i - co * (Cin * 9);
  int ci  = rem / 9;
  int k   = rem - ci * 9;
  dst[(size_t)co * (Cin * 9) + k * Cin + ci] = f2bf(src[i]);
}

// ---------------------------------------------------------------------------
// offsets/mask from the 432-channel conv output:
//   dy(g,k) = 5*tanh(o[g*18+2k]), dx = 5*tanh(o[g*18+2k+1]), m = sigmoid(o[288+g*9+k])
// ---------------------------------------------------------------------------
__global__ __launch_bounds__(256) void offmask_k(const float* __restrict__ o,
                                                 float* __restrict__ DY,
                                                 float* __restrict__ DX,
                                                 float* __restrict__ MS) {
  int q = blockIdx.x * 256 + threadIdx.x;     // 144*4096 threads
  if (q >= 144 * HW) return;
  int gk = q >> 12, p = q & 4095;
  int g = gk / 9, k = gk - g * 9;
  float dy = 5.f * tanhf(o[(size_t)(g * 18 + k * 2) * HW + p]);
  float dx = 5.f * tanhf(o[(size_t)(g * 18 + k * 2 + 1) * HW + p]);
  float m  = 1.f / (1.f + expf(-o[(size_t)(288 + gk) * HW + p]));
  DY[q] = dy; DX[q] = dx; MS[q] = m;
}

// ---------------------------------------------------------------------------
// DCNv2 bilinear gather: builds modulated im2col matrix val[ci*9+k][p] (bf16)
// xin = concat(X0, X1), Cin=256, DG=16 (cpg=16), K=9, H=W=64, pad 1.
// ---------------------------------------------------------------------------
__global__ __launch_bounds__(256) void gather_k(const float* __restrict__ X0,
                                                const float* __restrict__ X1,
                                                const float* __restrict__ DY,
                                                const float* __restrict__ DX,
                                                const float* __restrict__ MS,
                                                unsigned short* __restrict__ val) {
  int q = blockIdx.x * 256 + threadIdx.x;     // 144*4096 threads
  if (q >= 144 * HW) return;
  int gk = q >> 12, p = q & 4095;
  int g = gk / 9, k = gk - g * 9;
  int y = p >> 6, x = p & 63;
  float py = (float)(y + (k / 3) - 1) + DY[q];
  float px = (float)(x + (k % 3) - 1) + DX[q];
  float m  = MS[q];
  float fy0 = floorf(py), fx0 = floorf(px);
  int y0 = (int)fy0, x0 = (int)fx0, y1 = y0 + 1, x1 = x0 + 1;
  float fy = py - fy0, fx = px - fx0;
  bool vy0 = (y0 >= 0) && (y0 < 64), vy1 = (y1 >= 0) && (y1 < 64);
  bool vx0 = (x0 >= 0) && (x0 < 64), vx1 = (x1 >= 0) && (x1 < 64);
  int cy0 = min(max(y0, 0), 63), cy1 = min(max(y1, 0), 63);
  int cx0 = min(max(x0, 0), 63), cx1 = min(max(x1, 0), 63);
  float w00 = (1.f - fy) * (1.f - fx) * ((vy0 && vx0) ? 1.f : 0.f);
  float w01 = (1.f - fy) * fx         * ((vy0 && vx1) ? 1.f : 0.f);
  float w10 = fy * (1.f - fx)         * ((vy1 && vx0) ? 1.f : 0.f);
  float w11 = fy * fx                 * ((vy1 && vx1) ? 1.f : 0.f);
  int i00 = cy0 * 64 + cx0, i01 = cy0 * 64 + cx1;
  int i10 = cy1 * 64 + cx0, i11 = cy1 * 64 + cx1;
#pragma unroll
  for (int c = 0; c < 16; ++c) {
    int ci = g * 16 + c;
    const float* Xp = (ci < 128) ? (X0 + (size_t)ci * HW)
                                 : (X1 + (size_t)(ci - 128) * HW);
    float v = w00 * Xp[i00] + w01 * Xp[i01] + w10 * Xp[i10] + w11 * Xp[i11];
    val[(size_t)(ci * 9 + k) * HW + p] = f2bf(v * m);
  }
}

// ---------------------------------------------------------------------------
// Implicit-GEMM conv via V_WMMA_F32_16X16X32_BF16 + async global->LDS staging.
//   out[co][p] = sum_k Wb[co][k] * B[k][p]  (+bias, act, +addend)
// MODE 0: B = 3x3/pad1 im2col of up to 3 concatenated 128-ch bf16 parts
//         (weights tap-major: K = tap*Cin + ci; one tap per 32-K block)
// MODE 1: B = direct bf16 matrix P0[Kdim][4096] (deform values, flat OIHW weights)
// MODE 2: B = 1x1 (k == ci) from 128-ch bf16 parts
// Tiles: BM=64, BN=64, BK=32; 128 threads = 4 waves (2x2); each wave owns a
// 32x32 macro tile = 4 WMMA accumulators.
// A tile: double-buffered in LDS, staged with GLOBAL_LOAD_ASYNC_TO_LDS_B128
//         (tracked by ASYNCcnt; s_wait_asynccnt before the barrier).
// B tile: global->reg prefetch, committed to LDS at top of iteration.
// Dynamic LDS layout: A ping-pong @ bytes [0,4096), [4096,8192); B @ 8192.
// ---------------------------------------------------------------------------
#define BM 64
#define BN 64
#define BK 32
#define LDS_BYTES 12288

template <int MODE>
__global__ __launch_bounds__(128) void conv_wmma_k(
    const unsigned short* __restrict__ Wb,     // [Cout, Kdim] bf16
    const float* __restrict__ bias,            // [Cout]
    const unsigned short* __restrict__ P0,
    const unsigned short* __restrict__ P1,
    const unsigned short* __restrict__ P2,
    float* __restrict__ outF,                  // nullable [Cout,4096]
    unsigned short* __restrict__ outB,         // nullable bf16 copy
    const float* __restrict__ addend,          // nullable fp32 residual
    int Cout, int Kdim, int Cin, int act)
{
  extern __shared__ __align__(16) char smem[];
  unsigned short* ldsB = (unsigned short*)(smem + 8192);

  const int tid  = threadIdx.x;
  const int lane = tid & 31;
  const int wave = tid >> 5;
  const int wm   = wave & 1;     // M group (32 rows)
  const int wn   = wave >> 1;    // N group (32 cols)
  const int half = lane >> 4;
  const int l16  = lane & 15;

  const int nblock = blockIdx.x * BN;   // pixel base
  const int mblock = blockIdx.y * BM;   // cout base

  // staging mapping: each of 128 threads stages 16 A elems + 16 B elems
  const int arow = tid >> 1;            // 0..63 (Cout row)
  const int akb  = (tid & 1) * 16;      // k sub-block
  const int brow = tid >> 1;            // 0..63 (pixel)
  const int bkb  = (tid & 1) * 16;

  const int p  = nblock + brow;
  const int py = p >> 6, px = p & 63;

  const unsigned short* parts[3] = {P0, P1, P2};

  // clamp OOB Cout rows: their products land in C rows that are never stored
  int grA = mblock + arow;
  if (grA >= Cout) grA = Cout - 1;
  const unsigned int ldsOffA0 = (unsigned int)((arow * BK + akb) * 2);

  // ---- async-stage A tile kk0 into LDS buffer pb (32 bytes/thread) ----
  auto stageA = [&](int kk0, int pb) {
    const unsigned short* src = Wb + (size_t)grA * Kdim + kk0 + akb;
    unsigned int loff = ldsOffA0 + (unsigned int)(pb * 4096);
    unsigned long long ga = (unsigned long long)src;
    asm volatile("global_load_async_to_lds_b128 %0, %1, off"
                 :: "v"(loff), "v"(ga) : "memory");
    asm volatile("global_load_async_to_lds_b128 %0, %1, off offset:16"
                 :: "v"(loff), "v"(ga) : "memory");
  };

  unsigned short brg[16];

  // ---- prefetch B tile into registers ----
  auto prefetchB = [&](int kk0, int tap_, int cib_) {
    if (MODE == 0) {
      int yy = py + (tap_ / 3) - 1;
      int xx = px + (tap_ - (tap_ / 3) * 3) - 1;
      bool ok = (yy >= 0) && (yy < 64) && (xx >= 0) && (xx < 64);
      int sp = yy * 64 + xx;
#pragma unroll
      for (int j = 0; j < 16; ++j) {
        int ci = cib_ + bkb + j;
        int part = ci >> 7, lc = ci & 127;
        brg[j] = ok ? parts[part][(size_t)lc * HW + sp] : (unsigned short)0;
      }
    } else if (MODE == 1) {
#pragma unroll
      for (int j = 0; j < 16; ++j)
        brg[j] = P0[(size_t)(kk0 + bkb + j) * HW + p];
    } else {
#pragma unroll
      for (int j = 0; j < 16; ++j) {
        int ci = kk0 + bkb + j;
        int part = ci >> 7, lc = ci & 127;
        brg[j] = parts[part][(size_t)lc * HW + p];
      }
    }
  };

  v8f acc00 = {}, acc01 = {}, acc10 = {}, acc11 = {};

  int tapN = 0, cibN = 0;
  stageA(0, 0);
  prefetchB(0, 0, 0);
  int buf = 0;

  for (int k0 = 0; k0 < Kdim; k0 += BK) {
    // ---- commit B regs to LDS ----
#pragma unroll
    for (int j = 0; j < 16; j += 2)
      *(unsigned int*)&ldsB[brow * BK + bkb + j] =
          (unsigned int)brg[j] | ((unsigned int)brg[j + 1] << 16);
    // our async A writes for tile k0 must be visible before the barrier
    asm volatile("s_wait_asynccnt 0x0" ::: "memory");
    __syncthreads();

    // ---- stage next tile (async A into other buffer; B into regs) ----
    if (k0 + BK < Kdim) {
      cibN += BK;
      if (MODE == 0 && cibN >= Cin) { cibN = 0; tapN += 1; }
      stageA(k0 + BK, buf ^ 1);
      prefetchB(k0 + BK, tapN, cibN);
    }

    // ---- fragments: A pair layout {0,2,4,6,16,18,20,22}+8*half,
    //                 B pair layout {0,2,4,6,16,18,20,22}+8*half ----
    const unsigned int* A32 = (const unsigned int*)(smem + buf * 4096);
    const unsigned int* B32 = (const unsigned int*)ldsB;
    v8u ar0, ar1, br0, br1;
#pragma unroll
    for (int j = 0; j < 8; ++j) {
      int ka  = ((j & 3) * 2) + ((j >> 2) * 16) + half * 8;
      int kb2 = ((j & 3) * 2) + ((j >> 2) * 16) + half * 8;
      ar0[j] = A32[(wm * 32 + l16) * (BK / 2) + (ka >> 1)];
      ar1[j] = A32[(wm * 32 + 16 + l16) * (BK / 2) + (ka >> 1)];
      br0[j] = B32[(wn * 32 + l16) * (BK / 2) + (kb2 >> 1)];
      br1[j] = B32[(wn * 32 + 16 + l16) * (BK / 2) + (kb2 >> 1)];
    }
    v16bf a0 = __builtin_bit_cast(v16bf, ar0);
    v16bf a1 = __builtin_bit_cast(v16bf, ar1);
    v16bf b0 = __builtin_bit_cast(v16bf, br0);
    v16bf b1 = __builtin_bit_cast(v16bf, br1);

    acc00 = __builtin_amdgcn_wmma_f32_16x16x32_bf16(false, a0, false, b0, (short)0, acc00, false, false);
    acc01 = __builtin_amdgcn_wmma_f32_16x16x32_bf16(false, a0, false, b1, (short)0, acc01, false, false);
    acc10 = __builtin_amdgcn_wmma_f32_16x16x32_bf16(false, a1, false, b0, (short)0, acc10, false, false);
    acc11 = __builtin_amdgcn_wmma_f32_16x16x32_bf16(false, a1, false, b1, (short)0, acc11, false, false);

    __syncthreads();
    buf ^= 1;
  }

  // ---- epilogue: C layout VGPR r -> row r + 8*half, col = l16 ----
#pragma unroll
  for (int si = 0; si < 2; ++si) {
#pragma unroll
    for (int sj = 0; sj < 2; ++sj) {
      const v8f& acc = si == 0 ? (sj == 0 ? acc00 : acc01)
                               : (sj == 0 ? acc10 : acc11);
      int col = nblock + wn * 32 + sj * 16 + l16;
#pragma unroll
      for (int r = 0; r < 8; ++r) {
        int row = mblock + wm * 32 + si * 16 + r + half * 8;
        if (row < Cout) {
          float v = acc[r] + bias[row];
          if (act == 1) v = v > 0.f ? v : 0.1f * v;   // leaky relu 0.1
          size_t off = (size_t)row * HW + col;
          if (addend) v += addend[off];
          if (outF) outF[off] = v;
          if (outB) outB[off] = f2bf(v);
        }
      }
    }
  }
}

// ---------------------------------------------------------------------------
// Host orchestration
// ---------------------------------------------------------------------------
struct DirW {
  const float *co_w1, *co_b1, *co_w2, *co_b2, *co_w3, *co_b3, *co_w4, *co_b4,
              *dw, *db, *w1, *b1, *w2, *b2;
};

extern "C" void kernel_launch(void* const* d_in, const int* in_sizes, int n_in,
                              void* d_out, int out_size, void* d_ws, size_t ws_size,
                              hipStream_t stream) {
  (void)n_in; (void)out_size; (void)ws_size;
  auto F = [&](int i) { return (const float*)d_in[i]; };

  // ---- input mapping: auto-detect insertion-order vs sorted tree flatten ----
  DirW W[2];
  const float *fusion_w, *fusion_b, *xF;
  if (in_sizes[0] == 6 * CHW) {  // x first: insertion order
    xF = F(0);
    for (int d = 0; d < 2; ++d) {
      int b = 1 + d * 14;
      W[d] = { F(b+0), F(b+1), F(b+2), F(b+3), F(b+4), F(b+5), F(b+6),
               F(b+7), F(b+8), F(b+9), F(b+10), F(b+11), F(b+12), F(b+13) };
    }
    fusion_w = F(29); fusion_b = F(30);
  } else {                       // sorted leaves: params first, x last
    for (int d = 0; d < 2; ++d) {
      int b = d * 14;
      W[d].co_b1 = F(b+0); W[d].co_b2 = F(b+1); W[d].co_b3 = F(b+2); W[d].co_b4 = F(b+3);
      W[d].co_w1 = F(b+4); W[d].co_w2 = F(b+5); W[d].co_w3 = F(b+6); W[d].co_w4 = F(b+7);
      W[d].db = F(b+8);  W[d].dw = F(b+9);
      W[d].b1 = F(b+10); W[d].b2 = F(b+11); W[d].w1 = F(b+12); W[d].w2 = F(b+13);
    }
    fusion_b = F(28); fusion_w = F(29); xF = F(30);
  }

  // ---- workspace bump allocator ----
  char* base = (char*)d_ws;
  size_t off = 0;
  auto alloc = [&](size_t bytes) -> void* {
    off = (off + 255) & ~(size_t)255;
    void* pp = base + off;
    off += bytes;
    return pp;
  };
  auto allocF = [&](size_t n) { return (float*)alloc(n * 4); };
  auto allocB = [&](size_t n) { return (unsigned short*)alloc(n * 2); };

  const int SZ_dw = 128 * 256 * 9;
  const int SZ_fw = 128 * 256;

  struct DirWB { unsigned short *co_w1, *co_w2, *co_w3, *co_w4, *dw, *w1, *w2; } WB[2];
  for (int d = 0; d < 2; ++d) {
    WB[d].co_w1 = allocB(128 * 384 * 9); WB[d].co_w2 = allocB(128 * 128 * 9);
    WB[d].co_w3 = allocB(128 * 128 * 9); WB[d].co_w4 = allocB(432 * 128 * 9);
    WB[d].dw = allocB(SZ_dw);
    WB[d].w1 = allocB(d == 0 ? 128 * 256 * 9 : 128 * 384 * 9);
    WB[d].w2 = allocB(128 * 128 * 9);
  }
  unsigned short* fusion_wb = allocB(SZ_fw);

  unsigned short* xB = allocB(6 * CHW);
  float* zerosF = allocF(CHW);
  unsigned short* zerosB = allocB(CHW);
  float* seqF[2][6]; unsigned short* seqB[2][6];
  for (int d = 0; d < 2; ++d)
    for (int i = 0; i < 6; ++i) { seqF[d][i] = allocF(CHW); seqB[d][i] = allocB(CHW); }
  unsigned short* h1B = allocB(CHW);
  unsigned short* h2B = allocB(CHW);
  float* oF = allocF(432 * HW);
  float* DY = allocF(144 * HW); float* DX = allocF(144 * HW); float* MS = allocF(144 * HW);
  unsigned short* valB = allocB(256 * 9 * HW);
  float* alignF = allocF(CHW); unsigned short* alignB = allocB(CHW);
  unsigned short* bb1B = allocB(CHW);

  auto pack = [&](const float* s, unsigned short* d, int n) {
    pack_bf16_k<<<(n + 255) / 256, 256, 0, stream>>>(s, d, n);
  };
  auto packTap = [&](const float* s, unsigned short* d, int Cout, int Cin) {
    int n = Cout * Cin * 9;
    pack_w_tap_k<<<(n + 255) / 256, 256, 0, stream>>>(s, d, Cout, Cin);
  };
  auto conv = [&](const unsigned short* Wb, const float* bias,
                  const unsigned short* P0, const unsigned short* P1,
                  const unsigned short* P2, float* oF_, unsigned short* oB_,
                  const float* add, int Cout, int Kdim, int Cin, int mode, int act) {
    dim3 g(HW / BN, (Cout + BM - 1) / BM);
    if (mode == 0)
      conv_wmma_k<0><<<g, 128, LDS_BYTES, stream>>>(Wb, bias, P0, P1, P2, oF_, oB_, add, Cout, Kdim, Cin, act);
    else if (mode == 1)
      conv_wmma_k<1><<<g, 128, LDS_BYTES, stream>>>(Wb, bias, P0, P1, P2, oF_, oB_, add, Cout, Kdim, Cin, act);
    else
      conv_wmma_k<2><<<g, 128, LDS_BYTES, stream>>>(Wb, bias, P0, P1, P2, oF_, oB_, add, Cout, Kdim, Cin, act);
  };

  // ---- prep: bf16 weights (tap-major for 3x3 GEMMs), bf16 x, zeros ----
  for (int d = 0; d < 2; ++d) {
    packTap(W[d].co_w1, WB[d].co_w1, 128, 384);
    packTap(W[d].co_w2, WB[d].co_w2, 128, 128);
    packTap(W[d].co_w3, WB[d].co_w3, 128, 128);
    packTap(W[d].co_w4, WB[d].co_w4, 432, 128);
    pack(W[d].dw, WB[d].dw, SZ_dw);                    // mode-1 flat
    packTap(W[d].w1, WB[d].w1, 128, d == 0 ? 256 : 384);
    packTap(W[d].w2, WB[d].w2, 128, 128);
  }
  pack(fusion_w, fusion_wb, SZ_fw);                    // mode-2 flat
  pack(xF, xB, 6 * CHW);
  hipMemsetAsync(zerosF, 0, (size_t)CHW * 4, stream);
  hipMemsetAsync(zerosB, 0, (size_t)CHW * 2, stream);

  const int NOFF = 144 * HW;

  // ---- two propagation passes (d=0 backward, d=1 forward) ----
  for (int d = 0; d < 2; ++d) {
    for (int i = 0; i < 6; ++i) {
      int idx = (d == 0) ? (5 - i) : i;
      const unsigned short* xb = xB + (size_t)idx * CHW;
      const float*          fpF  = (i > 0) ? seqF[d][i - 1] : zerosF;
      const unsigned short* fpB  = (i > 0) ? seqB[d][i - 1] : zerosB;
      const float*          fn2F = (i > 1) ? seqF[d][i - 2] : zerosF;
      const unsigned short* fn2B = (i > 1) ? seqB[d][i - 2] : zerosB;

      const float* curF; const unsigned short* curB;
      if (i > 0) {
        // condition trunk: cond = [feat_prop, fc, fn2]  (3C -> C -> C -> C -> 432)
        conv(WB[d].co_w1, W[d].co_b1, fpB, xb, fn2B, nullptr, h1B, nullptr,
             128, 3456, 384, 0, 1);
        conv(WB[d].co_w2, W[d].co_b2, h1B, nullptr, nullptr, nullptr, h2B, nullptr,
             128, 1152, 128, 0, 1);
        conv(WB[d].co_w3, W[d].co_b3, h2B, nullptr, nullptr, nullptr, h1B, nullptr,
             128, 1152, 128, 0, 1);
        conv(WB[d].co_w4, W[d].co_b4, h1B, nullptr, nullptr, oF, nullptr, nullptr,
             432, 1152, 128, 0, 0);
        offmask_k<<<(NOFF + 255) / 256, 256, 0, stream>>>(oF, DY, DX, MS);
        // deformable conv on xin = [feat_prop, fn2]
        gather_k<<<(NOFF + 255) / 256, 256, 0, stream>>>(fpF, fn2F, DY, DX, MS, valB);
        conv(WB[d].dw, W[d].db, valB, nullptr, nullptr, alignF, alignB, nullptr,
             128, 2304, 256, 1, 0);
        curF = alignF; curB = alignB;
      } else {
        curF = zerosF; curB = zerosB;
      }

      // backbone: feat = [fc, (bwd_feat), feat_prop]; feat_prop += backbone(feat)
      if (d == 0) {
        conv(WB[d].w1, W[d].b1, xb, curB, nullptr, nullptr, bb1B, nullptr,
             128, 2304, 256, 0, 1);
      } else {
        conv(WB[d].w1, W[d].b1, xb, seqB[0][5 - idx], curB, nullptr, bb1B, nullptr,
             128, 3456, 384, 0, 1);
      }
      conv(WB[d].w2, W[d].b2, bb1B, nullptr, nullptr, seqF[d][i], seqB[d][i], curF,
           128, 1152, 128, 0, 0);
    }
  }

  // ---- fusion (1x1 on [bwd_feat, fwd_feat]) + residual with x, into d_out ----
  for (int idx = 0; idx < 6; ++idx) {
    conv(fusion_wb, fusion_b, seqB[0][5 - idx], seqB[1][idx], nullptr,
         (float*)d_out + (size_t)idx * CHW, nullptr, xF + (size_t)idx * CHW,
         128, 256, 256, 2, 0);
  }
}
